// LocalFeatureAggregation_30116310680159
// MI455X (gfx1250) — matrix operand order repaired and checked by reading, not compiled, not run on hardware
//
#include <hip/hip_runtime.h>
#include <hip/hip_bf16.h>
#include <math.h>

typedef __attribute__((ext_vector_type(16))) _Float16 v16h;
typedef __attribute__((ext_vector_type(8)))  _Float16 v8h;
typedef __attribute__((ext_vector_type(4)))  _Float16 v4h;
typedef __attribute__((ext_vector_type(8)))  float    v8f;

#define BB   2
#define NN   8192
#define KNB  16
#define CINC 128
#define DDIM 256
#define COUTC 256
#define NEGS 0.2f
#define EPSV 1e-5f

// ---- workspace byte offsets ----
#define WS_ATTN 0          // 8*16 tiles * 1024B = 131072
#define WS_OUT  131072     // 131072
#define WS_SC   262144     // 4*16 tiles * 1024B = 65536
#define WS_PRM  327680     // floats: relS[128] relT[128] outS[256] outT[256] scS[256] scT[256]

// ---- output element offsets (tuple concat) ----
#define O_XYZ  0
#define O_FEAT (2*8192*3)               // 49152
#define O_REL  (O_FEAT + 2*8192*256)    // 4243456
#define O_IDX  (O_REL + 2*8192*16*10)   // 6864896

// ---- LDS byte offsets ----
#define L_RELW 0            // 10*128 f32 = 5120
#define L_PRM  5120         // relS[128] relT[128] f32 = 1024
#define L_FEAT 6144         // 16 waves * 16 rows * 272 halves * 2B = 139264
#define L_SC   145408       // 16 rows * 136 halves * 2B = 4352
#define L_POOL 149760       // 16 rows * 272 halves * 2B = 8704
#define L_TOTAL 158464

// ================= prep: fold BN into per-channel scale/shift =================
__global__ void lfa_prep_params(const float* rel_b, const float* rel_g, const float* rel_be,
                                const float* rel_m, const float* rel_v,
                                const float* out_b, const float* out_g, const float* out_be,
                                const float* out_m, const float* out_v,
                                const float* sc_b, const float* sc_g, const float* sc_be,
                                const float* sc_m, const float* sc_v,
                                float* prm) {
  int c = threadIdx.x;
  if (c < 128) {
    float S = rel_g[c] * rsqrtf(rel_v[c] + EPSV);
    prm[c]        = S;
    prm[128 + c]  = (rel_b[c] - rel_m[c]) * S + rel_be[c];
  }
  if (c < 256) {
    float S  = out_g[c] * rsqrtf(out_v[c] + EPSV);
    prm[256 + c]  = S;
    prm[512 + c]  = (out_b[c] - out_m[c]) * S + out_be[c];
    float S2 = sc_g[c] * rsqrtf(sc_v[c] + EPSV);
    prm[768 + c]  = S2;
    prm[1024 + c] = (sc_b[c] - sc_m[c]) * S2 + sc_be[c];
  }
}

// ========== prep: swizzle row-major fp32 weights -> WMMA f16 B-operand ==========
// B operand (16x16x32 f16): lane l holds column n = l%16; VGPR v<4 holds K pair
// (8h+2v, 8h+2v+1), VGPR v>=4 holds (8+8h+2v, 9+8h+2v), h = lane half.
// Tile storage (1024B): chunk0 (VGPR0-3) then chunk1 (VGPR4-7), each lane's 16B
// contiguous so the consumer does two global_load_b128 per tile.
__device__ inline void swz_one(const float* __restrict__ W, _Float16* __restrict__ dst, int local) {
  int tile = local >> 8, rem = local & 255;
  int chunk = rem >> 7, lrem = rem & 127;
  int lane = lrem >> 2, j = lrem & 3;
  int kb = tile >> 4, nT = tile & 15;
  int V = chunk * 4 + j;
  int k = kb * 32 + 2 * V + (chunk << 3) + ((lane >> 4) << 3);
  int ncol = nT * 16 + (lane & 15);
  dst[2 * local]     = (_Float16)W[(size_t)k * 256 + ncol];
  dst[2 * local + 1] = (_Float16)W[(size_t)(k + 1) * 256 + ncol];
}

__global__ void lfa_prep_swz(const float* __restrict__ attnW, const float* __restrict__ outW,
                             const float* __restrict__ scW, _Float16* __restrict__ wsh) {
  int t = blockIdx.x * blockDim.x + threadIdx.x;
  if (t < 32768)        swz_one(attnW, wsh,                 t);
  else if (t < 65536)   swz_one(outW,  wsh + WS_OUT / 2,    t - 32768);
  else if (t < 81920)   swz_one(scW,   wsh + WS_SC / 2,     t - 65536);
}

// ================= pass-through outputs =================
__global__ void lfa_copy(const float* __restrict__ xyz, const float* __restrict__ relf,
                         const int* __restrict__ nidx, float* __restrict__ out) {
  size_t t = (size_t)blockIdx.x * blockDim.x + threadIdx.x;
  size_t stride = (size_t)gridDim.x * blockDim.x;
  for (size_t i = t; i < (size_t)2 * 8192 * 3; i += stride)        out[O_XYZ + i] = xyz[i];
  for (size_t i = t; i < (size_t)2 * 8192 * 16 * 10; i += stride)  out[O_REL + i] = relf[i];
  unsigned* uo = (unsigned*)out;
  const unsigned* ui = (const unsigned*)nidx;
  for (size_t i = t; i < (size_t)2 * 8192 * 16; i += stride)       uo[O_IDX + i] = ui[i];
}

// ================= main fused kernel =================
// 1024 blocks x 512 threads (16 wave32). Wave w handles point p = blk*16 + w.
__global__ __launch_bounds__(512)
void lfa_main(const float* __restrict__ feature,
              const float* __restrict__ relf,
              const int*   __restrict__ nidx,
              const float* __restrict__ relW,
              const _Float16* __restrict__ wsh,
              const float* __restrict__ prm,
              float* __restrict__ out) {
  extern __shared__ char smem[];
  float* relW_l = (float*)(smem + L_RELW);
  float* relS_l = (float*)(smem + L_PRM);
  float* relT_l = relS_l + 128;
  _Float16* sL  = (_Float16*)(smem + L_SC);
  _Float16* pL  = (_Float16*)(smem + L_POOL);

  const int tid = threadIdx.x;
  const int w = tid >> 5;
  const int l = tid & 31;
  const int lhi = l >> 4;          // 0 or 1 (lane half)
  const int lm = l & 15;

  // cooperative fill: rel_W + fused rel BN params into LDS
  for (int i = tid; i < 1280; i += 512) relW_l[i] = relW[i];
  if (tid < 128) { relS_l[tid] = prm[tid]; relT_l[tid] = prm[128 + tid]; }
  __syncthreads();

  const int p = blockIdx.x * 16 + w;
  const int b = p >> 13;
  const int n = p & 8191;
  const size_t bn = (size_t)b * NN + n;
  const float* fbase = feature + (size_t)b * NN * CINC;

  _Float16* fL = (_Float16*)(smem + L_FEAT) + w * (16 * 272);

  // ---- gather K=16 neighbor rows -> feat[:, 0:128] (f16) ----
  int myidx = nidx[bn * KNB + lm];
  for (int m = 0; m < 16; ++m) {
    int r = __shfl(myidx, m, 32);
    const float4 g = *(const float4*)(fbase + (size_t)r * CINC + l * 4);
    v4h h; h[0] = (_Float16)g.x; h[1] = (_Float16)g.y; h[2] = (_Float16)g.z; h[3] = (_Float16)g.w;
    *(v4h*)(fL + m * 272 + l * 4) = h;
  }

  // ---- rel MLP (16x10 @ 10x128) + fused BN + leaky -> feat[:, 128:256] ----
  const float* xb = relf + bn * KNB * 10;
  for (int m = 0; m < 16; ++m) {
    const float* x = xb + m * 10;
    float xv[10];
#pragma unroll
    for (int t = 0; t < 10; ++t) xv[t] = x[t];
    float y0 = 0.f, y1 = 0.f, y2 = 0.f, y3 = 0.f;
#pragma unroll
    for (int t = 0; t < 10; ++t) {
      const float4 wv = *(const float4*)(relW_l + t * 128 + l * 4);
      y0 += xv[t] * wv.x; y1 += xv[t] * wv.y; y2 += xv[t] * wv.z; y3 += xv[t] * wv.w;
    }
    const int c0 = l * 4;
    const float4 s4 = *(const float4*)(relS_l + c0);
    const float4 t4 = *(const float4*)(relT_l + c0);
    float r0 = y0 * s4.x + t4.x; r0 = r0 < 0.f ? NEGS * r0 : r0;
    float r1 = y1 * s4.y + t4.y; r1 = r1 < 0.f ? NEGS * r1 : r1;
    float r2 = y2 * s4.z + t4.z; r2 = r2 < 0.f ? NEGS * r2 : r2;
    float r3 = y3 * s4.w + t4.w; r3 = r3 < 0.f ? NEGS * r3 : r3;
    v4h h; h[0] = (_Float16)r0; h[1] = (_Float16)r1; h[2] = (_Float16)r2; h[3] = (_Float16)r3;
    *(v4h*)(fL + m * 272 + 128 + c0) = h;
  }

  // ---- this point's own feature row -> scL (for shortcut GEMM A) ----
  {
    const float4 g = *(const float4*)(fbase + (size_t)n * CINC + l * 4);
    v4h h; h[0] = (_Float16)g.x; h[1] = (_Float16)g.y; h[2] = (_Float16)g.z; h[3] = (_Float16)g.w;
    *(v4h*)(sL + w * 136 + l * 4) = h;
  }

  // ---- load full 16x256 A matrix into 64 VGPRs (A layout: 16-bit 16x32) ----
  v16h A[8];
#pragma unroll
  for (int kb = 0; kb < 8; ++kb) {
    const _Float16* base = fL + lm * 272 + kb * 32 + (lhi << 3);
    v8h lo = *(const v8h*)base;
    v8h hi = *(const v8h*)(base + 16);
    A[kb] = __builtin_shufflevector(lo, hi, 0,1,2,3,4,5,6,7,8,9,10,11,12,13,14,15);
  }

  // ---- attention logits: 16 N-tiles x 8 k-steps WMMA, fused softmax+pooling ----
  const _Float16* attnB = wsh + WS_ATTN / 2;
  for (int nT = 0; nT < 16; ++nT) {
    v8f acc = {0.f, 0.f, 0.f, 0.f, 0.f, 0.f, 0.f, 0.f};
#pragma unroll
    for (int kb = 0; kb < 8; ++kb) {
      const _Float16* tb = attnB + (size_t)(kb * 16 + nT) * 512 + l * 8;
      v8h blo = *(const v8h*)tb;
      v8h bhi = *(const v8h*)(tb + 256);
      v16h Bm = __builtin_shufflevector(blo, bhi, 0,1,2,3,4,5,6,7,8,9,10,11,12,13,14,15);
      acc = __builtin_amdgcn_wmma_f32_16x16x32_f16(false, A[kb], false, Bm, (short)0, acc, false, false);
    }
    // softmax over the 16 neighbors (M dim): column n lives in lanes n and n+16
    float mx = acc[0];
#pragma unroll
    for (int v = 1; v < 8; ++v) mx = fmaxf(mx, acc[v]);
    mx = fmaxf(mx, __shfl_xor(mx, 16, 32));
    float e[8], s = 0.f;
#pragma unroll
    for (int v = 0; v < 8; ++v) { e[v] = __expf(acc[v] - mx); s += e[v]; }
    s += __shfl_xor(s, 16, 32);
    float ps = 0.f;
    const int col = nT * 16 + lm;
#pragma unroll
    for (int v = 0; v < 8; ++v) {
      int m = v + (lhi << 3);
      ps += e[v] * (float)fL[m * 272 + col];
    }
    ps += __shfl_xor(ps, 16, 32);
    float pooled = ps / s;
    if (l < 16) pL[w * 272 + col] = (_Float16)pooled;
  }
  __syncthreads();

  // ---- phase B: 16 points batched; wave w owns output N-tile nT=w ----
  v16h AM[8];
#pragma unroll
  for (int kb = 0; kb < 8; ++kb) {
    const _Float16* base = pL + lm * 272 + kb * 32 + (lhi << 3);
    v8h lo = *(const v8h*)base;
    v8h hi = *(const v8h*)(base + 16);
    AM[kb] = __builtin_shufflevector(lo, hi, 0,1,2,3,4,5,6,7,8,9,10,11,12,13,14,15);
  }
  v8f accM = {0.f, 0.f, 0.f, 0.f, 0.f, 0.f, 0.f, 0.f};
  const _Float16* outB = wsh + WS_OUT / 2;
#pragma unroll
  for (int kb = 0; kb < 8; ++kb) {
    const _Float16* tb = outB + (size_t)(kb * 16 + w) * 512 + l * 8;
    v8h blo = *(const v8h*)tb;
    v8h bhi = *(const v8h*)(tb + 256);
    v16h Bm = __builtin_shufflevector(blo, bhi, 0,1,2,3,4,5,6,7,8,9,10,11,12,13,14,15);
    accM = __builtin_amdgcn_wmma_f32_16x16x32_f16(false, AM[kb], false, Bm, (short)0, accM, false, false);
  }

  v8f accS = {0.f, 0.f, 0.f, 0.f, 0.f, 0.f, 0.f, 0.f};
  const _Float16* scB = wsh + WS_SC / 2;
#pragma unroll
  for (int kb = 0; kb < 4; ++kb) {
    const _Float16* base = sL + lm * 136 + kb * 32 + (lhi << 3);
    v8h lo = *(const v8h*)base;
    v8h hi = *(const v8h*)(base + 16);
    v16h Am = __builtin_shufflevector(lo, hi, 0,1,2,3,4,5,6,7,8,9,10,11,12,13,14,15);
    const _Float16* tb = scB + (size_t)(kb * 16 + w) * 512 + l * 8;
    v8h blo = *(const v8h*)tb;
    v8h bhi = *(const v8h*)(tb + 256);
    v16h Bm = __builtin_shufflevector(blo, bhi, 0,1,2,3,4,5,6,7,8,9,10,11,12,13,14,15);
    accS = __builtin_amdgcn_wmma_f32_16x16x32_f16(false, Am, false, Bm, (short)0, accS, false, false);
  }

  // ---- epilogue: fused BN on both paths, add, leaky, store ----
  const int c = w * 16 + lm;
  const float oS = prm[256 + c], oT = prm[512 + c];
  const float sS = prm[768 + c], sT = prm[1024 + c];
#pragma unroll
  for (int v = 0; v < 8; ++v) {
    int m = v + (lhi << 3);
    float r = accM[v] * oS + oT + accS[v] * sS + sT;
    r = r < 0.f ? NEGS * r : r;
    out[O_FEAT + (size_t)(blockIdx.x * 16 + m) * COUTC + c] = r;
  }
}

extern "C" void kernel_launch(void* const* d_in, const int* in_sizes, int n_in,
                              void* d_out, int out_size, void* d_ws, size_t ws_size,
                              hipStream_t stream) {
  const float* xyz     = (const float*)d_in[0];
  const float* feature = (const float*)d_in[1];
  const float* relf    = (const float*)d_in[2];
  const int*   nidx    = (const int*)d_in[3];
  const float* rel_W   = (const float*)d_in[4];
  const float* rel_b   = (const float*)d_in[5];
  const float* rel_g   = (const float*)d_in[6];
  const float* rel_be  = (const float*)d_in[7];
  const float* rel_m   = (const float*)d_in[8];
  const float* rel_v   = (const float*)d_in[9];
  const float* attn_W  = (const float*)d_in[10];
  const float* out_W   = (const float*)d_in[11];
  const float* out_b   = (const float*)d_in[12];
  const float* out_g   = (const float*)d_in[13];
  const float* out_be  = (const float*)d_in[14];
  const float* out_m   = (const float*)d_in[15];
  const float* out_v   = (const float*)d_in[16];
  const float* sc_W    = (const float*)d_in[17];
  const float* sc_b    = (const float*)d_in[18];
  const float* sc_g    = (const float*)d_in[19];
  const float* sc_be   = (const float*)d_in[20];
  const float* sc_m    = (const float*)d_in[21];
  const float* sc_v    = (const float*)d_in[22];

  _Float16* wsh = (_Float16*)d_ws;
  float* prm = (float*)((char*)d_ws + WS_PRM);
  float* outp = (float*)d_out;

  lfa_prep_params<<<1, 256, 0, stream>>>(rel_b, rel_g, rel_be, rel_m, rel_v,
                                         out_b, out_g, out_be, out_m, out_v,
                                         sc_b, sc_g, sc_be, sc_m, sc_v, prm);
  lfa_prep_swz<<<320, 256, 0, stream>>>(attn_W, out_W, sc_W, wsh);
  lfa_copy<<<2048, 256, 0, stream>>>(xyz, relf, nidx, outp);
  lfa_main<<<1024, 512, L_TOTAL, stream>>>(feature, relf, nidx, rel_W, wsh, prm, outp);
}
